// VGAE_53085795778670
// MI455X (gfx1250) — matrix operand order repaired and checked by reading, not compiled, not run on hardware
//
#include <hip/hip_runtime.h>
#include <cstdint>
#include <cstddef>

// ---------------- types ----------------
typedef __attribute__((ext_vector_type(16))) __bf16        v16bf;
typedef __attribute__((ext_vector_type(8)))  unsigned int  v8u;
typedef __attribute__((ext_vector_type(8)))  float         v8f;

#define VN 10000
#define VE 320000
#define VF 512
#define VH 256
#define VO 64

// ---------------- helpers ----------------
__device__ __forceinline__ unsigned short f2bf(float f) {
  unsigned int u = __builtin_bit_cast(unsigned int, f);
  unsigned int r = u + 0x7fffu + ((u >> 16) & 1u);   // round-to-nearest-even
  return (unsigned short)(r >> 16);
}

__device__ __forceinline__ v8f wmma_bf16(v16bf a, v16bf b, v8f c) {
  // (neg_a, A, neg_b, B, c_mod, C, reuse_a, reuse_b)
  return __builtin_amdgcn_wmma_f32_16x16x32_bf16(false, a, false, b, (short)0, c,
                                                 false, false);
}

// A-matrix fragment (16x32 bf16). base -> (row0,k0) of row-major ushort LDS.
// ISA 7.12.2: lanes 0-15 M=0..15 (K 0..7 in v0..3, 16..23 in v4..7),
//             lanes 16-31 same M with K+8.
__device__ __forceinline__ v16bf frag_A(const unsigned short* base, int stride, int lane) {
  int m = lane & 15, hi = lane >> 4;
  const unsigned short* rp = base + m * stride;
  v8u u;
#pragma unroll
  for (int v = 0; v < 8; ++v) {
    int k = (v < 4) ? (2 * v + hi * 8) : (16 + 2 * (v - 4) + hi * 8);
    u[v] = *(const unsigned int*)(rp + k);
  }
  return __builtin_bit_cast(v16bf, u);
}

// B-matrix fragment (32x16 bf16). base -> (col0,k0) of an LDS array stored
// [col][k] (K-major per column). lanes 0-15: N=lane, K 0..7 in v0..3, 16..23 in v4..7;
// lanes 16-31: same N with K+8.
__device__ __forceinline__ v16bf frag_B(const unsigned short* base, int stride, int lane) {
  int n = lane & 15, hi = lane >> 4;
  const unsigned short* rp = base + n * stride;
  v8u u;
#pragma unroll
  for (int v = 0; v < 8; ++v) u[v] = *(const unsigned int*)(rp + ((v < 4) ? (2 * v + hi * 8) : (16 + 2 * (v - 4) + hi * 8)));
  return __builtin_bit_cast(v16bf, u);
}

__device__ __forceinline__ float sigmoidf(float x) {
  return 1.0f / (1.0f + __expf(-x));
}

// ---------------- elementwise kernels ----------------
__global__ void cast_bf16_kernel(const float* __restrict__ in,
                                 unsigned short* __restrict__ out, int n) {
  int i = blockIdx.x * blockDim.x + threadIdx.x;
  if (i < n) out[i] = f2bf(in[i]);
}

__global__ void relu_cast_bf16_kernel(const float* __restrict__ in,
                                      unsigned short* __restrict__ out, int n) {
  int i = blockIdx.x * blockDim.x + threadIdx.x;
  if (i < n) out[i] = f2bf(fmaxf(in[i], 0.0f));
}

__global__ void zero_kernel(float* __restrict__ p, int n) {
  int i = blockIdx.x * blockDim.x + threadIdx.x;
  if (i < n) p[i] = 0.0f;
}

// Z = noise * exp(logstd) + mean  -> bf16
__global__ void reparam_kernel(const float* __restrict__ noise,
                               const float* __restrict__ mean,
                               const float* __restrict__ logstd,
                               unsigned short* __restrict__ zb, int n) {
  int i = blockIdx.x * blockDim.x + threadIdx.x;
  if (i < n) zb[i] = f2bf(noise[i] * __expf(logstd[i]) + mean[i]);
}

// ---------------- edge-parallel spmm (segment_sum via f32 atomics) ----------------
__global__ void spmm_kernel(const int* __restrict__ src, const int* __restrict__ dst,
                            const float* __restrict__ w, const float* __restrict__ x,
                            float* __restrict__ out, int E, int D4) {
  int t = blockIdx.x * blockDim.x + threadIdx.x;
  if (t >= E * D4) return;
  int e = t / D4;
  int c = (t - e * D4) * 4;
  int D = D4 * 4;
  float wt = w[e];
  const float4 xv = *(const float4*)(x + (size_t)src[e] * D + c);
  float* o = out + (size_t)dst[e] * D + c;
  atomicAdd(o + 0, wt * xv.x);
  atomicAdd(o + 1, wt * xv.y);
  atomicAdd(o + 2, wt * xv.z);
  atomicAdd(o + 3, wt * xv.w);
}

// ---------------- generic bf16 WMMA GEMM: C = A[MxK] @ B[KxNc] + bias ----------------
// block tile 64x64, 8 waves (4x2), each wave 16x32; K stepped by 32.
__global__ __launch_bounds__(256) void gemm_bf16_kernel(
    const unsigned short* __restrict__ A, const unsigned short* __restrict__ B,
    const float* __restrict__ bias, float* __restrict__ C, int M, int K, int Nc) {
  __shared__ unsigned short As[64][40];  // [m][k]
  __shared__ unsigned short Bt[64][40];  // [n][k]  (transposed stage)
  int tid = threadIdx.x;
  int lane = tid & 31, w = tid >> 5;
  int mblk = blockIdx.y * 64, nblk = blockIdx.x * 64;
  int m_off = (w & 3) * 16, n_off = (w >> 2) * 32;

  v8f acc0 = {}; v8f acc1 = {};

  for (int k0 = 0; k0 < K; k0 += 32) {
    {  // stage A: 64 rows x 32 k, one uint4 per thread
      int row = tid >> 2, seg = (tid & 3) * 8;
      uint4 val = make_uint4(0u, 0u, 0u, 0u);
      int gr = mblk + row;
      if (gr < M) val = *(const uint4*)(A + (size_t)gr * K + k0 + seg);
      *(uint4*)&As[row][seg] = val;
    }
    {  // stage B transposed: 32 k-rows x 64 n, unpack uint4 -> 8 scattered u16
      int kr = tid >> 3, seg = (tid & 7) * 8;
      uint4 val = *(const uint4*)(B + (size_t)(k0 + kr) * Nc + nblk + seg);
      unsigned int p[4] = {val.x, val.y, val.z, val.w};
#pragma unroll
      for (int j = 0; j < 4; ++j) {
        Bt[seg + 2 * j + 0][kr] = (unsigned short)(p[j] & 0xffffu);
        Bt[seg + 2 * j + 1][kr] = (unsigned short)(p[j] >> 16);
      }
    }
    __syncthreads();
    v16bf af = frag_A(&As[m_off][0], 40, lane);
    v16bf b0 = frag_B(&Bt[n_off][0], 40, lane);
    v16bf b1 = frag_B(&Bt[n_off + 16][0], 40, lane);
    acc0 = wmma_bf16(af, b0, acc0);
    acc1 = wmma_bf16(af, b1, acc1);
    __syncthreads();
  }

  int n_l = lane & 15, hi = lane >> 4;
  float bv0 = bias[nblk + n_off + n_l];
  float bv1 = bias[nblk + n_off + 16 + n_l];
#pragma unroll
  for (int r = 0; r < 8; ++r) {
    int gr = mblk + m_off + hi * 8 + r;   // C layout: lanes<16 -> M=r, lanes>=16 -> M=8+r
    if (gr < M) {
      C[(size_t)gr * Nc + nblk + n_off + n_l]      = acc0[r] + bv0;
      C[(size_t)gr * Nc + nblk + n_off + 16 + n_l] = acc1[r] + bv1;
    }
  }
}

// ---------------- adj_rec = sigmoid(Z @ Z^T), Z bf16 [N][64] ----------------
// block tile 128x128, 8 waves (2x4), each wave 64x32 (4x2 accum tiles), K=64.
// Interior blocks (97.5% of grid) take a guard-free fast path; the 400MB output
// is streamed with non-temporal stores so it does not thrash L2.
__global__ __launch_bounds__(256) void zzt_sigmoid_kernel(
    const unsigned short* __restrict__ Zb, float* __restrict__ out, int N) {
  __shared__ unsigned short Am[128][72];  // output-row Z rows   [m][k]
  __shared__ unsigned short Bn[128][72];  // output-col Z rows   [n][k]
  int tid = threadIdx.x, lane = tid & 31, w = tid >> 5;
  int mblk = blockIdx.y * 128, nblk = blockIdx.x * 128;
  bool interior = (mblk + 128 <= N) && (nblk + 128 <= N);

  {  // stage: 2 threads per row, each 32 bf16 (64B)
    int row = tid >> 1, half = (tid & 1) * 32;
    int gr = mblk + row;
    int gc = nblk + row;
    if (interior) {
#pragma unroll
      for (int q = 0; q < 4; ++q) {
        uint4 va = *(const uint4*)(Zb + (size_t)gr * VO + half + q * 8);
        uint4 vb = *(const uint4*)(Zb + (size_t)gc * VO + half + q * 8);
        *(uint4*)&Am[row][half + q * 8] = va;
        *(uint4*)&Bn[row][half + q * 8] = vb;
      }
    } else {
#pragma unroll
      for (int q = 0; q < 4; ++q) {
        uint4 va = make_uint4(0u, 0u, 0u, 0u);
        uint4 vb = make_uint4(0u, 0u, 0u, 0u);
        if (gr < N) va = *(const uint4*)(Zb + (size_t)gr * VO + half + q * 8);
        if (gc < N) vb = *(const uint4*)(Zb + (size_t)gc * VO + half + q * 8);
        *(uint4*)&Am[row][half + q * 8] = va;
        *(uint4*)&Bn[row][half + q * 8] = vb;
      }
    }
  }
  __syncthreads();

  int m_off = (w >> 2) * 64;  // 0 or 64
  int n_off = (w & 3) * 32;   // 0,32,64,96

  v16bf a[4][2];
#pragma unroll
  for (int mt = 0; mt < 4; ++mt)
#pragma unroll
    for (int kk = 0; kk < 2; ++kk)
      a[mt][kk] = frag_A(&Am[m_off + mt * 16][kk * 32], 72, lane);

  v16bf b[2][2];
#pragma unroll
  for (int nt = 0; nt < 2; ++nt)
#pragma unroll
    for (int kk = 0; kk < 2; ++kk)
      b[nt][kk] = frag_B(&Bn[n_off + nt * 16][kk * 32], 72, lane);

  v8f acc[4][2];
#pragma unroll
  for (int mt = 0; mt < 4; ++mt)
#pragma unroll
    for (int nt = 0; nt < 2; ++nt) {
      v8f c = {};
      c = wmma_bf16(a[mt][0], b[nt][0], c);
      c = wmma_bf16(a[mt][1], b[nt][1], c);
      acc[mt][nt] = c;
    }

  int n_l = lane & 15, hi = lane >> 4;
  if (interior) {
    // guard-free streaming stores
#pragma unroll
    for (int mt = 0; mt < 4; ++mt)
#pragma unroll
      for (int nt = 0; nt < 2; ++nt) {
        size_t col = (size_t)(nblk + n_off + nt * 16 + n_l);
        size_t rbase = (size_t)(mblk + m_off + mt * 16 + hi * 8);
#pragma unroll
        for (int r = 0; r < 8; ++r) {
          __builtin_nontemporal_store(sigmoidf(acc[mt][nt][r]),
                                      out + (rbase + r) * (size_t)N + col);
        }
      }
  } else {
#pragma unroll
    for (int mt = 0; mt < 4; ++mt)
#pragma unroll
      for (int nt = 0; nt < 2; ++nt) {
        int col = nblk + n_off + nt * 16 + n_l;
#pragma unroll
        for (int r = 0; r < 8; ++r) {
          int row = mblk + m_off + mt * 16 + hi * 8 + r;
          if (row < N && col < N)
            __builtin_nontemporal_store(sigmoidf(acc[mt][nt][r]),
                                        out + (size_t)row * N + col);
        }
      }
  }
}

// ---------------- launch ----------------
extern "C" void kernel_launch(void* const* d_in, const int* in_sizes, int n_in,
                              void* d_out, int out_size, void* d_ws, size_t ws_size,
                              hipStream_t stream) {
  (void)in_sizes; (void)n_in; (void)out_size; (void)ws_size;
  const float* features    = (const float*)d_in[0];
  const int*   edge_src    = (const int*)d_in[1];
  const int*   edge_dst    = (const int*)d_in[2];
  const float* edge_weight = (const float*)d_in[3];
  const float* noise       = (const float*)d_in[4];
  const float* W1          = (const float*)d_in[5];
  const float* b1          = (const float*)d_in[6];
  const float* Wm          = (const float*)d_in[7];
  const float* bm          = (const float*)d_in[8];
  const float* Wl          = (const float*)d_in[9];
  const float* bl          = (const float*)d_in[10];

  float* out        = (float*)d_out;
  float* mean_out   = out + (size_t)VN * VN;
  float* logstd_out = mean_out + (size_t)VN * VO;

  // workspace carve-up (256B aligned)
  char* ws = (char*)d_ws;
  size_t off = 0;
  auto carve = [&](size_t bytes) -> void* {
    void* p = ws + off;
    off = (off + bytes + 255) & ~(size_t)255;
    return p;
  };
  unsigned short* featb = (unsigned short*)carve((size_t)VN * VF * 2);
  unsigned short* W1b   = (unsigned short*)carve((size_t)VF * VH * 2);
  unsigned short* Wmb   = (unsigned short*)carve((size_t)VH * VO * 2);
  unsigned short* Wlb   = (unsigned short*)carve((size_t)VH * VO * 2);
  unsigned short* hb    = (unsigned short*)carve((size_t)VN * VH * 2);
  unsigned short* Zb    = (unsigned short*)carve((size_t)VN * VO * 2);
  float* t1 = (float*)carve((size_t)VN * VH * 4);
  float* g1 = (float*)carve((size_t)VN * VH * 4);
  float* tm = (float*)carve((size_t)VN * VO * 4);
  float* tl = (float*)carve((size_t)VN * VO * 4);

  const int T = 256;
  auto blks = [](long long n, int t) { return (unsigned)((n + t - 1) / t); };

  // 1) cast weights/features to bf16
  cast_bf16_kernel<<<blks((long long)VN * VF, T), T, 0, stream>>>(features, featb, VN * VF);
  cast_bf16_kernel<<<blks((long long)VF * VH, T), T, 0, stream>>>(W1, W1b, VF * VH);
  cast_bf16_kernel<<<blks((long long)VH * VO, T), T, 0, stream>>>(Wm, Wmb, VH * VO);
  cast_bf16_kernel<<<blks((long long)VH * VO, T), T, 0, stream>>>(Wl, Wlb, VH * VO);

  // 2) t1 = features @ W1 + b1
  gemm_bf16_kernel<<<dim3(VH / 64, (VN + 63) / 64), T, 0, stream>>>(
      featb, W1b, b1, t1, VN, VF, VH);

  // 3) g1 = spmm(t1)
  zero_kernel<<<blks((long long)VN * VH, T), T, 0, stream>>>(g1, VN * VH);
  spmm_kernel<<<blks((long long)VE * (VH / 4), T), T, 0, stream>>>(
      edge_src, edge_dst, edge_weight, t1, g1, VE, VH / 4);

  // 4) h = bf16(relu(g1))
  relu_cast_bf16_kernel<<<blks((long long)VN * VH, T), T, 0, stream>>>(g1, hb, VN * VH);

  // 5) heads: tm = h@Wm+bm, tl = h@Wl+bl
  gemm_bf16_kernel<<<dim3(VO / 64, (VN + 63) / 64), T, 0, stream>>>(
      hb, Wmb, bm, tm, VN, VH, VO);
  gemm_bf16_kernel<<<dim3(VO / 64, (VN + 63) / 64), T, 0, stream>>>(
      hb, Wlb, bl, tl, VN, VH, VO);

  // 6) mean/logstd = spmm(tm/tl) straight into d_out
  zero_kernel<<<blks((long long)VN * VO, T), T, 0, stream>>>(mean_out, VN * VO);
  zero_kernel<<<blks((long long)VN * VO, T), T, 0, stream>>>(logstd_out, VN * VO);
  spmm_kernel<<<blks((long long)VE * (VO / 4), T), T, 0, stream>>>(
      edge_src, edge_dst, edge_weight, tm, mean_out, VE, VO / 4);
  spmm_kernel<<<blks((long long)VE * (VO / 4), T), T, 0, stream>>>(
      edge_src, edge_dst, edge_weight, tl, logstd_out, VE, VO / 4);

  // 7) Z (bf16) = noise * exp(logstd) + mean
  reparam_kernel<<<blks((long long)VN * VO, T), T, 0, stream>>>(
      noise, mean_out, logstd_out, Zb, VN * VO);

  // 8) adj_rec = sigmoid(Z @ Z^T)
  zzt_sigmoid_kernel<<<dim3((VN + 127) / 128, (VN + 127) / 128), T, 0, stream>>>(
      Zb, out, VN);
}
